// GemmaAttentionWeightsOnly_10943576670974
// MI455X (gfx1250) — matrix operand order repaired and checked by reading, not compiled, not run on hardware
//
#include <hip/hip_runtime.h>
#include <cstdint>

typedef __attribute__((ext_vector_type(16))) _Float16 v16h;
typedef __attribute__((ext_vector_type(8)))  float    v8f;

union Frag16 { v16h v; _Float16 h[16]; uint32_t u[8]; };
union AccF   { v8f  v; float f[8]; };

#define B_    2
#define L_    2048
#define D_    3584
#define HD_   256
#define NQ_   16
#define NKV_  8
#define NH_   24               // NQ + NKV heads packed in one proj buffer
#define NPROJ_ (NH_*HD_)       // 6144

#if __has_builtin(__builtin_amdgcn_global_load_async_to_lds_b128) && \
    __has_builtin(__builtin_amdgcn_s_wait_asynccnt)
#define HAVE_ASYNC_LDS 1
#else
#define HAVE_ASYNC_LDS 0
#endif

#if HAVE_ASYNC_LDS
// Builtin expects GNU-vector int4 pointers in AS1 (global) / AS3 (LDS).
typedef int v4i_gnu __attribute__((vector_size(16)));
typedef __attribute__((address_space(1))) v4i_gnu* as1_v4i;
typedef __attribute__((address_space(3))) v4i_gnu* as3_v4i;
__device__ __forceinline__ void async_copy16(const void* g, void* l) {
  // per-lane 16B copy, global -> LDS, tracked by ASYNCcnt
  __builtin_amdgcn_global_load_async_to_lds_b128(
      (as1_v4i)(uintptr_t)g, (as3_v4i)(uintptr_t)l, 0, 0);
}
#endif

// ---------------------------------------------------------------------------
// Kernel 1: fused Q/K projection GEMM  (M=4096, N=6144, K=3584), f16 WMMA
// qk[m, n] = sum_k hidden[m,k] * w[n,k], stored f16 as [B,L,NH,HD]
// Data path: async global->LDS DMA (double-buffered), f32->f16 at frag build.
// ---------------------------------------------------------------------------
__global__ __launch_bounds__(256) void proj_kernel(
    const float* __restrict__ hidden, const float* __restrict__ wq,
    const float* __restrict__ wk, _Float16* __restrict__ qk)
{
  const int tid  = threadIdx.x;
  const int lane = tid & 31;
  const int wave = tid >> 5;
  const int m0 = blockIdx.y * 64;
  const int n0 = blockIdx.x * 128;
  const int wm = (wave & 1) * 32;    // wave M offset: 0/32
  const int wn = (wave >> 1) * 32;   // wave N offset: 0..96

  // N tile is entirely inside wq or wk (4096 % 128 == 0)
  const float* wbase = (n0 < NQ_*HD_) ? (wq + (size_t)n0 * D_)
                                      : (wk + (size_t)(n0 - NQ_*HD_) * D_);

  const int mloc = lane & 15;
  const int kbA  = (lane < 16) ? 0 : 16;    // A frag: half-wave K split 0-15/16-31
  const int kbB  = (lane < 16) ? 0 : 16;    // B frag: half-wave K split
  AccF acc[2][2] = {};

#if HAVE_ASYNC_LDS
  // f32 staging tiles, double buffered; padded stride 36 dwords (conflict-free)
  __shared__ float As32[2][64][36];
  __shared__ float Bs32[2][128][36];

  auto issue_stage = [&](int buf, int k0) {
    // A: 64x32 f32 = 512 x 16B, 2 per thread; B: 128x32 = 1024 x 16B, 4/thread
    for (int t = 0; t < 2; ++t) {
      const int i = tid + t*256, row = i >> 3, seg = i & 7;
      async_copy16(hidden + (size_t)(m0 + row) * D_ + k0 + seg*4,
                   &As32[buf][row][seg*4]);
    }
    for (int t = 0; t < 4; ++t) {
      const int i = tid + t*256, row = i >> 3, seg = i & 7;
      async_copy16(wbase + (size_t)row * D_ + k0 + seg*4,
                   &Bs32[buf][row][seg*4]);
    }
  };

  issue_stage(0, 0);                         // prologue: stage 0 -> buf 0
  for (int kt = 0; kt < 112; ++kt) {         // D/32 = 112 K-steps
    const int buf = kt & 1;
    if (kt < 111) {
      issue_stage(buf ^ 1, (kt + 1) * 32);   // overlap next stage with compute
      __builtin_amdgcn_s_wait_asynccnt(6);   // this wave's previous 6 done
    } else {
      __builtin_amdgcn_s_wait_asynccnt(0);
    }
    __syncthreads();                         // all waves' DMA visible

    Frag16 afr[2], bfr[2];
    for (int t = 0; t < 2; ++t) {
      const float* ar = As32[buf][wm + t*16 + mloc];
      for (int i = 0; i < 4; ++i) {
        afr[t].h[2*i]     = (_Float16)ar[kbA + 2*i];
        afr[t].h[2*i+1]   = (_Float16)ar[kbA + 2*i + 1];
        afr[t].h[8+2*i]   = (_Float16)ar[8 + kbA + 2*i];
        afr[t].h[8+2*i+1] = (_Float16)ar[8 + kbA + 2*i + 1];
      }
      const float* br = Bs32[buf][wn + t*16 + mloc];
      for (int i = 0; i < 8; ++i) {
        bfr[t].h[2*i]   = (_Float16)br[kbB + 2*i];
        bfr[t].h[2*i+1] = (_Float16)br[kbB + 2*i + 1];
      }
    }
    for (int mi = 0; mi < 2; ++mi)
      for (int nj = 0; nj < 2; ++nj)
        acc[mi][nj].v = __builtin_amdgcn_wmma_f32_16x16x32_f16(
            false, afr[mi].v, false, bfr[nj].v, (short)0, acc[mi][nj].v,
            false, false);
    __syncthreads();                         // buf reusable next iteration
  }
#else
  // Fallback: vectorized float4 loads, register double-buffer
  __shared__ _Float16 As[64][40];
  __shared__ _Float16 Bs[128][40];
  float4 ra[2], rb[4];

  auto load_regs = [&](int k0) {
    for (int t = 0; t < 2; ++t) {
      const int i = tid + t*256, row = i >> 3, seg = i & 7;
      ra[t] = *(const float4*)(hidden + (size_t)(m0 + row) * D_ + k0 + seg*4);
    }
    for (int t = 0; t < 4; ++t) {
      const int i = tid + t*256, row = i >> 3, seg = i & 7;
      rb[t] = *(const float4*)(wbase + (size_t)row * D_ + k0 + seg*4);
    }
  };
  auto store_lds = [&]() {
    for (int t = 0; t < 2; ++t) {
      const int i = tid + t*256, row = i >> 3, seg = i & 7;
      As[row][seg*4+0] = (_Float16)ra[t].x; As[row][seg*4+1] = (_Float16)ra[t].y;
      As[row][seg*4+2] = (_Float16)ra[t].z; As[row][seg*4+3] = (_Float16)ra[t].w;
    }
    for (int t = 0; t < 4; ++t) {
      const int i = tid + t*256, row = i >> 3, seg = i & 7;
      Bs[row][seg*4+0] = (_Float16)rb[t].x; Bs[row][seg*4+1] = (_Float16)rb[t].y;
      Bs[row][seg*4+2] = (_Float16)rb[t].z; Bs[row][seg*4+3] = (_Float16)rb[t].w;
    }
  };

  load_regs(0);
  for (int kt = 0; kt < 112; ++kt) {
    store_lds();
    __syncthreads();
    if (kt < 111) load_regs((kt + 1) * 32);  // overlap with WMMA below

    Frag16 afr[2], bfr[2];
    for (int t = 0; t < 2; ++t) {
      const _Float16* ar = As[wm + t*16 + mloc];
      for (int i = 0; i < 4; ++i) {
        afr[t].h[2*i]     = ar[kbA + 2*i];
        afr[t].h[2*i+1]   = ar[kbA + 2*i + 1];
        afr[t].h[8+2*i]   = ar[8 + kbA + 2*i];
        afr[t].h[8+2*i+1] = ar[8 + kbA + 2*i + 1];
      }
      const _Float16* br = Bs[wn + t*16 + mloc];
      for (int i = 0; i < 8; ++i) {
        bfr[t].h[2*i]   = br[kbB + 2*i];
        bfr[t].h[2*i+1] = br[kbB + 2*i + 1];
      }
    }
    for (int mi = 0; mi < 2; ++mi)
      for (int nj = 0; nj < 2; ++nj)
        acc[mi][nj].v = __builtin_amdgcn_wmma_f32_16x16x32_f16(
            false, afr[mi].v, false, bfr[nj].v, (short)0, acc[mi][nj].v,
            false, false);
    __syncthreads();
  }
#endif

  const int rowadd = (lane >> 4) * 8;   // C/D layout: lanes 16-31 are M+8
  const int coladd = lane & 15;
  for (int mi = 0; mi < 2; ++mi)
    for (int nj = 0; nj < 2; ++nj)
      for (int r = 0; r < 8; ++r) {
        const int m = m0 + wm + mi*16 + r + rowadd;
        const int n = n0 + wn + nj*16 + coladd;
        qk[(size_t)m * NPROJ_ + n] = (_Float16)acc[mi][nj].f[r];
      }
}

// ---------------------------------------------------------------------------
// Kernel 2: in-place RoPE on the f16 Q/K buffer. One thread per (d, d+128) pair.
// ---------------------------------------------------------------------------
__global__ __launch_bounds__(256) void rope_kernel(
    const float* __restrict__ cosb, const float* __restrict__ sinb,
    _Float16* __restrict__ qk)
{
  size_t idx = (size_t)blockIdx.x * 256 + threadIdx.x;
  const int d = (int)(idx & 127);
  size_t t = idx >> 7;
  const int head = (int)(t % NH_);  t /= NH_;
  const int l    = (int)(t % L_);
  const int b    = (int)(t / L_);

  const size_t cs = ((size_t)b * L_ + l) * HD_;
  const float c1 = cosb[cs + d],       s1 = sinb[cs + d];
  const float c2 = cosb[cs + d + 128], s2 = sinb[cs + d + 128];

  const size_t base = (((size_t)b * L_ + l) * NH_ + head) * HD_;
  const float x1 = (float)qk[base + d];
  const float x2 = (float)qk[base + d + 128];
  qk[base + d]       = (_Float16)(x1 * c1 - x2 * s1);   // x*cos + rot_half(x)*sin
  qk[base + d + 128] = (_Float16)(x2 * c2 + x1 * s2);
}

// ---------------------------------------------------------------------------
// Kernel 3: scores = softmax(tanh((Q Kᵀ)·scale/50)·50 + mask)
// One workgroup per (b, q-head, 16-query block). Full 16x2048 score rows held
// in 139 KB dynamic LDS (CDNA5: 320 KB/WGP), softmax fused, single HBM write.
// ---------------------------------------------------------------------------
__global__ __launch_bounds__(256) void scores_kernel(
    const _Float16* __restrict__ qk, const float* __restrict__ mask,
    float* __restrict__ out)
{
  extern __shared__ char smem[];
  float*    sc = (float*)smem;                         // [16][2048] scores
  _Float16* qs = (_Float16*)(smem + 16*2048*4);        // [16][264] Q tile (padded)
  const int QS = 264;

  const int tid  = threadIdx.x;
  const int lane = tid & 31;
  const int wave = tid >> 5;
  const int bid  = blockIdx.x;
  const int qb  = bid & 127;                // L/16 query blocks
  const int qh  = (bid >> 7) & 15;
  const int b   = bid >> 11;
  const int l0  = qb * 16;
  const int kvh = qh >> 1;                  // GQA: expand = 2

  // stage Q tile 16x256 into LDS (coalesced)
  for (int i = tid; i < 16*256; i += 256) {
    int r = i >> 8, c = i & 255;
    qs[r*QS + c] = qk[(((size_t)b*L_ + l0 + r)*NH_ + qh)*HD_ + c];
  }
  __syncthreads();

  // preload A fragments for all 8 K-steps (K = 256 = 8 x 32)
  const int mloc = lane & 15;
  const int kbA  = (lane < 16) ? 0 : 16;
  const int kbB  = (lane < 16) ? 0 : 16;
  Frag16 afr[8];
  for (int ks = 0; ks < 8; ++ks) {
    const _Float16* qrow = qs + mloc*QS + ks*32;
    for (int i = 0; i < 4; ++i) {
      afr[ks].h[2*i]     = qrow[kbA + 2*i];
      afr[ks].h[2*i+1]   = qrow[kbA + 2*i + 1];
      afr[ks].h[8+2*i]   = qrow[8 + kbA + 2*i];
      afr[ks].h[8+2*i+1] = qrow[8 + kbA + 2*i + 1];
    }
  }

  const float scl     = 0.0625f;        // 256^-0.5
  const float inv_cap = 1.0f / 50.0f;
  const size_t kheadbase = ((size_t)b*L_*NH_ + (NQ_ + kvh)) * HD_;

  // each wave handles key tiles wave, wave+8, ... (128 tiles total)
  for (int tile = wave; tile < L_/16; tile += 8) {
    const int key = tile*16 + mloc;     // B fragment: lane%16 = key (column)
    const _Float16* krow = qk + kheadbase + (size_t)key * NH_ * HD_;
    if (tile + 8 < L_/16) {             // prefetch this lane's next key row
      const _Float16* knext = krow + (size_t)128 * NH_ * HD_;
      __builtin_prefetch(knext, 0, 1);
      __builtin_prefetch(knext + 128, 0, 1);
    }
    const uint32_t* kp = (const uint32_t*)krow;
    AccF accu = {};
    for (int ks = 0; ks < 8; ++ks) {
      Frag16 bfr;
      const uint32_t* kk = kp + (ks*32 + kbB)/2;   // 16 contiguous halves
      for (int i = 0; i < 8; ++i) bfr.u[i] = kk[i];
      accu.v = __builtin_amdgcn_wmma_f32_16x16x32_f16(
          false, afr[ks].v, false, bfr.v, (short)0, accu.v, false, false);
    }
    // epilogue: scale, tanh softcap, +mask -> LDS row buffer
    const int col    = tile*16 + mloc;
    const int rowadd = (lane >> 4) * 8;
    for (int r = 0; r < 8; ++r) {
      const int row = r + rowadd;
      float s = accu.f[r] * scl;
      s = tanhf(s * inv_cap) * 50.0f;
      s += mask[((size_t)b*L_ + (l0 + row))*L_ + col];
      sc[row*2048 + col] = s;
    }
  }
  __syncthreads();

  // softmax: wave w owns rows 2w and 2w+1; wave32 shuffle reductions
  for (int rr = 0; rr < 2; ++rr) {
    const int row = wave*2 + rr;
    float* srow = sc + row*2048;
    float mx = -3.4e38f;
    for (int c = lane; c < L_; c += 32) mx = fmaxf(mx, srow[c]);
    for (int off = 16; off > 0; off >>= 1) mx = fmaxf(mx, __shfl_xor(mx, off, 32));
    float sum = 0.f;
    for (int c = lane; c < L_; c += 32) {
      float e = __expf(srow[c] - mx);
      srow[c] = e;
      sum += e;
    }
    for (int off = 16; off > 0; off >>= 1) sum += __shfl_xor(sum, off, 32);
    const float inv = 1.0f / sum;
    float* orow = out + (((size_t)b*NQ_ + qh)*L_ + l0 + row) * L_;
    for (int c = lane; c < L_; c += 32) orow[c] = srow[c] * inv;
  }
}

// ---------------------------------------------------------------------------
extern "C" void kernel_launch(void* const* d_in, const int* in_sizes, int n_in,
                              void* d_out, int out_size, void* d_ws, size_t ws_size,
                              hipStream_t stream) {
  (void)in_sizes; (void)n_in; (void)out_size; (void)ws_size;
  const float* hidden = (const float*)d_in[0];
  const float* cosb   = (const float*)d_in[1];
  const float* sinb   = (const float*)d_in[2];
  const float* mask   = (const float*)d_in[3];
  const float* wq     = (const float*)d_in[4];
  const float* wk     = (const float*)d_in[5];
  _Float16* qk = (_Float16*)d_ws;            // [B,L,24,256] f16 = 48 MiB

  const int kSmemBytes = 16*2048*4 + 16*264*2;   // 139,520 B (< 320 KB/WGP)
  (void)hipFuncSetAttribute(reinterpret_cast<const void*>(scores_kernel),
                            hipFuncAttributeMaxDynamicSharedMemorySize, kSmemBytes);

  proj_kernel<<<dim3(NPROJ_/128, (B_*L_)/64), 256, 0, stream>>>(hidden, wq, wk, qk);
  rope_kernel<<<dim3((B_*L_*NH_*(HD_/2))/256), 256, 0, stream>>>(cosb, sinb, qk);
  scores_kernel<<<dim3(B_*NQ_*(L_/16)), 256, kSmemBytes, stream>>>(qk, mask, (float*)d_out);
}
